// DeformableConv2dGivenOffset_30253749633352
// MI455X (gfx1250) — matrix operand order repaired and checked by reading, not compiled, not run on hardware
//
#include <hip/hip_runtime.h>

typedef __attribute__((ext_vector_type(16))) _Float16 v16h;
typedef __attribute__((ext_vector_type(8)))  _Float16 v8h;
typedef __attribute__((ext_vector_type(8)))  float    v8f;

#define K_    3
#define KK_   9          // K*K taps
#define G_    8          // groups
#define CPG   8          // channels per group
#define OPG   8          // outputs per group
#define B_    4
#define CIN   64
#define OC_   64
#define H_    256
#define W_    256
#define HW_   (H_ * W_)
#define KDIM  72         // KK_ * CPG reduction length
#define KPAD  96         // padded to 3 * 32 for wmma f16 k-steps
#define TILE  16         // pixels per block (one row segment)
#define WPG   (OPG * CPG * KK_)   // 576 weights per group, contiguous

__global__ __launch_bounds__(256)
void dcn2_wmma_kernel(const float* __restrict__ x,
                      const float* __restrict__ offset,
                      const float* __restrict__ mask,
                      const float* __restrict__ weight,
                      const float* __restrict__ bias,
                      float* __restrict__ out)
{
    // per-wave private tiles, both n/pixel-major with contiguous K (f16)
    __shared__ __attribute__((aligned(32))) _Float16 ldsA[G_][TILE * KPAD]; // samples
    __shared__ __attribute__((aligned(32))) _Float16 ldsB[G_][TILE * KPAD]; // weights

    const int lane = threadIdx.x & 31;
    const int g    = threadIdx.x >> 5;          // wave id == group id
    const int x0   = blockIdx.x * TILE;
    const int y    = blockIdx.y;
    const int b    = blockIdx.z;

    const v8h z8 = {};

    // ---- zero A-tile K padding (k = 72..95, 24 f16 = 3x v8h per row) ----
    if (lane < TILE) {
        _Float16* pz = &ldsA[g][lane * KPAD + KDIM];
        *(v8h*)(pz + 0)  = z8;
        *(v8h*)(pz + 8)  = z8;
        *(v8h*)(pz + 16) = z8;
    }

    // ---- zero entire B tile (rows n>=8 and K pad stay zero) ----
    #pragma unroll
    for (int idx = 0; idx < (TILE * KPAD) / 8; idx += 32) {
        *(v8h*)&ldsB[g][(idx + lane) * 8] = z8;
    }

    // ---- stage group weights: coalesced global read, scatter to [n][k] ----
    // weight block for group g is contiguous: 576 floats at weight + g*576
    {
        const float* wsrc = weight + g * WPG;
        #pragma unroll
        for (int it = 0; it < WPG / 32; ++it) {       // 18 iterations
            const int idx = it * 32 + lane;
            const float wv = wsrc[idx];               // coalesced b32
            const int n    = idx / (CPG * KK_);       // output channel in group
            const int r    = idx - n * (CPG * KK_);
            const int cc   = r / KK_;                 // input channel in group
            const int tap  = r - cc * KK_;            // tap index t
            ldsB[g][n * KPAD + tap * CPG + cc] = (_Float16)wv;
        }
    }

    // ---- bilinear gather (f32) + sigmoid mask -> LDS A tile (f16) ----
    const int    obase = b * 2 * (G_ * KK_) * HW_;
    const int    mbase = b * (G_ * KK_) * HW_;
    const float* xg    = x + (size_t)(b * CIN + g * CPG) * HW_;

    for (int i = lane; i < TILE * KK_; i += 32) {
        const int p   = i & 15;                 // pixel within tile
        const int t   = i >> 4;                 // tap 0..8
        const int xx  = x0 + p;
        const int pos = y * W_ + xx;
        // reference flips the combined (group, tap) axis of offset
        const int ch  = (G_ * KK_ - 1) - (g * KK_ + t);

        const float dy = offset[obase + ch * HW_ + pos];
        const float dx = offset[obase + (G_ * KK_ + ch) * HW_ + pos];
        const float mr = mask[mbase + (g * KK_ + t) * HW_ + pos];
        const float mv = 1.0f / (1.0f + __expf(-mr));

        const int ki = t / K_, kj = t % K_;
        const float py = (float)y  - 1.0f + (float)ki + dy;   // PAD = 1
        const float px = (float)xx - 1.0f + (float)kj + dx;
        const float fy = floorf(py), fx = floorf(px);
        const float wy = py - fy,    wx = px - fx;
        const int y0i = (int)fy, x0i = (int)fx;
        const int y1i = y0i + 1, x1i = x0i + 1;

        const float vy0 = (y0i >= 0 && y0i < H_) ? 1.0f : 0.0f;
        const float vy1 = (y1i >= 0 && y1i < H_) ? 1.0f : 0.0f;
        const float vx0 = (x0i >= 0 && x0i < W_) ? 1.0f : 0.0f;
        const float vx1 = (x1i >= 0 && x1i < W_) ? 1.0f : 0.0f;
        const int yc0 = min(max(y0i, 0), H_ - 1);
        const int yc1 = min(max(y1i, 0), H_ - 1);
        const int xc0 = min(max(x0i, 0), W_ - 1);
        const int xc1 = min(max(x1i, 0), W_ - 1);

        const float w00 = (1.0f - wy) * (1.0f - wx) * vy0 * vx0;
        const float w01 = (1.0f - wy) * wx          * vy0 * vx1;
        const float w10 = wy          * (1.0f - wx) * vy1 * vx0;
        const float w11 = wy          * wx          * vy1 * vx1;

        const int o00 = yc0 * W_ + xc0;
        const int o01 = yc0 * W_ + xc1;
        const int o10 = yc1 * W_ + xc0;
        const int o11 = yc1 * W_ + xc1;

        v8h sv;
        #pragma unroll
        for (int c = 0; c < CPG; ++c) {
            const float* pc = xg + c * HW_;
            const float s = w00 * pc[o00] + w01 * pc[o01]
                          + w10 * pc[o10] + w11 * pc[o11];
            sv[c] = (_Float16)(s * mv);
        }
        // one ds_store_b128: [pixel][t*8 .. t*8+7]
        *(v8h*)&ldsA[g][p * KPAD + t * CPG] = sv;
    }

    __syncthreads();

    // ---- WMMA: D[16px x 16n] += A[16px x 96k] * B[96k x 16n], 3 k-steps ----
    const int n  = lane & 15;        // A-matrix row (pixel) / B-matrix col for this lane
    const int hi = (lane >= 16);
    v8f acc = {};

    #pragma unroll
    for (int ks = 0; ks < 3; ++ks) {
        // A fragment: elems 0..15 <-> k = base..base+15 contiguous
        const _Float16* ap = &ldsA[g][n * KPAD + (hi ? 16 : 0) + 32 * ks];
        const v8h alo = *(const v8h*)(ap + 0);
        const v8h ahi = *(const v8h*)(ap + 8);
        const v16h a = __builtin_shufflevector(alo, ahi,
                           0, 1, 2, 3, 4, 5, 6, 7, 8, 9, 10, 11, 12, 13, 14, 15);
        // B fragment: elems 0..15 <-> k = base..base+15 contiguous
        const v16h bf = *(const v16h*)&ldsB[g][n * KPAD + (hi ? 16 : 0) + 32 * ks];

        acc = __builtin_amdgcn_wmma_f32_16x16x32_f16(
                  false, a, false, bf, (short)0, acc, false, false);
    }

    // ---- store D: lane%16 = n (output channel), VGPR r + 8*hi = pixel ----
    if (n < OPG) {
        const int oc = g * OPG + n;
        const float bv = bias[oc];
        float* op = out + ((size_t)(b * OC_ + oc) * H_ + y) * W_ + x0 + (hi ? 8 : 0);
        #pragma unroll
        for (int r = 0; r < 8; ++r) op[r] = acc[r] + bv;
    }
}

extern "C" void kernel_launch(void* const* d_in, const int* in_sizes, int n_in,
                              void* d_out, int out_size, void* d_ws, size_t ws_size,
                              hipStream_t stream) {
    const float* x      = (const float*)d_in[0];
    const float* offset = (const float*)d_in[1];
    const float* mask   = (const float*)d_in[2];
    const float* weight = (const float*)d_in[3];
    const float* bias   = (const float*)d_in[4];
    float* out = (float*)d_out;

    dim3 grid(W_ / TILE, H_, B_);   // 16 x 256 x 4 = 16384 blocks
    dcn2_wmma_kernel<<<grid, 256, 0, stream>>>(x, offset, mask, weight, bias, out);
}
